// Assigner_14422500180529
// MI455X (gfx1250) — compile-verified, matrix operand order checked by reading, not run on hardware
//
#include <hip/hip_runtime.h>
#include <hip/hip_bf16.h>
#include <math.h>

#define NA 8400
#define NB 64
#define BS 16
#define NC 80
#define TOPK 13

typedef __attribute__((ext_vector_type(16))) _Float16 v16h;
typedef __attribute__((ext_vector_type(8)))  float    v8f;

// ---------------------------------------------------------------------------
// K1: per (b, anchor) x 64 GTs -> align_metric, overlaps (CIoU) matrices.
// GT tables staged into LDS with CDNA5 async global->LDS loads.
// ---------------------------------------------------------------------------
__global__ void k1_metrics(const float* __restrict__ pd_scores,
                           const float* __restrict__ pd_bboxes,
                           const float* __restrict__ anc_points,
                           const int*   __restrict__ gt_labels,
                           const float* __restrict__ gt_bboxes,
                           const float* __restrict__ mask_gt,
                           float* __restrict__ align_ws,
                           float* __restrict__ ovl_ws) {
    __shared__ float s_gt[NB * 4];
    __shared__ int   s_lab[NB];
    __shared__ float s_mg[NB];
    __shared__ float s_atan[NB];
    __shared__ float s_w1h1[NB];

    const int b   = blockIdx.x;
    const int tid = threadIdx.x;

    // --- CDNA5 async global->LDS staging (tracked by ASYNCcnt) ---
    if (tid < NB) {  // 64 lanes x 16B = gt_bboxes[b] (1 KB)
        const float* g = gt_bboxes + ((size_t)b * NB + tid) * 4;
        uint32_t l = (uint32_t)(uintptr_t)(&s_gt[tid * 4]);
        asm volatile("global_load_async_to_lds_b128 %0, %1, off"
                     :: "v"(l), "v"(g) : "memory");
    }
    if (tid < NB / 4) {  // 16 lanes x 16B each for labels + mask
        const int* gl = gt_labels + (size_t)b * NB + tid * 4;
        uint32_t l0 = (uint32_t)(uintptr_t)(&s_lab[tid * 4]);
        asm volatile("global_load_async_to_lds_b128 %0, %1, off"
                     :: "v"(l0), "v"(gl) : "memory");
        const float* gm = mask_gt + (size_t)b * NB + tid * 4;
        uint32_t l1 = (uint32_t)(uintptr_t)(&s_mg[tid * 4]);
        asm volatile("global_load_async_to_lds_b128 %0, %1, off"
                     :: "v"(l1), "v"(gm) : "memory");
    }
    asm volatile("s_wait_asynccnt 0" ::: "memory");
    __syncthreads();

    // per-GT constants: atan(w1/h1), w1*h1 (hoists all transcendentals)
    if (tid < NB) {
        float w1 = s_gt[4 * tid + 2] - s_gt[4 * tid + 0];
        float h1 = s_gt[4 * tid + 3] - s_gt[4 * tid + 1] + 1e-7f;
        s_atan[tid] = atanf(w1 / h1);
        s_w1h1[tid] = w1 * h1;
    }
    __syncthreads();

    const int a = blockIdx.y * blockDim.x + tid;
    if (a >= NA) return;

    const float ax = anc_points[2 * a + 0];
    const float ay = anc_points[2 * a + 1];
    const float* pb = pd_bboxes + ((size_t)b * NA + a) * 4;
    const float px1 = pb[0], py1 = pb[1], px2 = pb[2], py2 = pb[3];
    const float w2 = px2 - px1, h2 = py2 - py1 + 1e-7f;
    const float atp  = atanf(w2 / h2);
    const float w2h2 = w2 * h2;
    const float cxp = px1 + px2, cyp = py1 + py2;
    const float* srow = pd_scores + ((size_t)b * NA + a) * NC;
    __builtin_prefetch(srow, 0, 3);  // global_prefetch_b8
    const float K4PI2 = 0.40528473456935109f;  // 4/pi^2

    for (int n = 0; n < NB; ++n) {
        const float gx1 = s_gt[4 * n + 0], gy1 = s_gt[4 * n + 1];
        const float gx2 = s_gt[4 * n + 2], gy2 = s_gt[4 * n + 3];
        float alignv = 0.f, ovlv = 0.f;
        float dmin = fminf(fminf(ax - gx1, ay - gy1), fminf(gx2 - ax, gy2 - ay));
        if (dmin > 1e-9f && s_mg[n] > 0.f) {
            float sc = srow[s_lab[n]];
            float iw = fmaxf(fminf(gx2, px2) - fmaxf(gx1, px1), 0.f);
            float ih = fmaxf(fminf(gy2, py2) - fmaxf(gy1, py1), 0.f);
            float inter = iw * ih;
            float uni = s_w1h1[n] + w2h2 - inter + 1e-7f;
            float iou = inter / uni;
            float cw = fmaxf(gx2, px2) - fminf(gx1, px1);
            float ch = fmaxf(gy2, py2) - fminf(gy1, py1);
            float c2 = cw * cw + ch * ch + 1e-7f;
            float dx = cxp - gx1 - gx2, dy = cyp - gy1 - gy2;
            float rho2 = (dx * dx + dy * dy) * 0.25f;
            float ad = atp - s_atan[n];
            float v = K4PI2 * ad * ad;
            float alpha = v / (v - iou + 1.f + 1e-7f);
            float ciou = iou - (rho2 / c2 + v * alpha);
            ovlv = fmaxf(ciou, 0.f);
            float o2 = ovlv * ovlv;
            alignv = sc * (o2 * o2 * o2);  // score^1 * ovl^6
        }
        const size_t idx = ((size_t)b * NB + n) * NA + a;
        align_ws[idx] = alignv;   // coalesced across threads per n
        ovl_ws[idx]   = ovlv;
    }
}

// ---------------------------------------------------------------------------
// K2: per (b, gt): top-13 of align row (LDS-resident), emit mask_pos row.
// ---------------------------------------------------------------------------
__global__ void k2_topk(const float* __restrict__ anc_points,
                        const float* __restrict__ gt_bboxes,
                        const float* __restrict__ mask_gt,
                        const float* __restrict__ align_ws,
                        float* __restrict__ mpos_ws) {
    __shared__ float row[NA];
    __shared__ float sval[256];
    __shared__ int   sidx[256];
    __shared__ int   sel[TOPK];

    const int b = blockIdx.x, n = blockIdx.y, tid = threadIdx.x;
    const size_t base = ((size_t)b * NB + n) * NA;
    const float mg = mask_gt[b * NB + n];  // uniform per block
    const float gx1 = gt_bboxes[((size_t)b * NB + n) * 4 + 0];
    const float gy1 = gt_bboxes[((size_t)b * NB + n) * 4 + 1];
    const float gx2 = gt_bboxes[((size_t)b * NB + n) * 4 + 2];
    const float gy2 = gt_bboxes[((size_t)b * NB + n) * 4 + 3];

    int nsel = 0;
    if (mg > 0.f) {
        for (int a = tid; a < NA; a += 256) row[a] = align_ws[base + a];
        __syncthreads();
        for (int it = 0; it < TOPK; ++it) {
            float bv = -1e30f; int bi = 0;
            for (int a = tid; a < NA; a += 256) {
                float v = row[a];
                if (v > bv) { bv = v; bi = a; }  // first-index tiebreak
            }
            sval[tid] = bv; sidx[tid] = bi;
            __syncthreads();
            for (int s = 128; s > 0; s >>= 1) {
                if (tid < s) {
                    if (sval[tid + s] > sval[tid] ||
                        (sval[tid + s] == sval[tid] && sidx[tid + s] < sidx[tid])) {
                        sval[tid] = sval[tid + s]; sidx[tid] = sidx[tid + s];
                    }
                }
                __syncthreads();
            }
            if (tid == 0) { sel[it] = sidx[0]; row[sidx[0]] = -1e30f; }
            __syncthreads();
        }
        nsel = TOPK;
    }
    __syncthreads();
    // single coalesced pass: mask_pos = topk && in_gts && mask_gt
    for (int a = tid; a < NA; a += 256) {
        float v = 0.f;
        for (int k = 0; k < nsel; ++k) {
            if (a == sel[k]) {
                float axx = anc_points[2 * a], ayy = anc_points[2 * a + 1];
                float dmin = fminf(fminf(axx - gx1, ayy - gy1),
                                   fminf(gx2 - axx, gy2 - ayy));
                if (dmin > 1e-9f) v = 1.f;
            }
        }
        mpos_ws[base + a] = v;
    }
}

// ---------------------------------------------------------------------------
// K3: fg = sum_n mask_pos via V_WMMA_F32_16X16X32_F16 (A = ones 16x32,
// B = mask chunk 32x16, two K-chunks accumulate the 64-deep column sum).
// Then resolve multi-assignment, emit fg_mask / target_gt_idx / target_bboxes.
// ---------------------------------------------------------------------------
__global__ void k3_resolve(const float* __restrict__ ovl_ws,
                           float* __restrict__ mpos_ws,
                           const float* __restrict__ gt_bboxes,
                           float* __restrict__ fg_ws,
                           int*   __restrict__ tgi_ws,
                           float* __restrict__ out_tb,
                           float* __restrict__ out_fg,
                           float* __restrict__ out_tgi) {
    const int b    = blockIdx.x;
    const int tid  = threadIdx.x;
    const int lane = tid & 31;
    const int wave = tid >> 5;
    const int a0   = blockIdx.y * 128 + wave * 16;  // 16 anchors per wave
    const int col  = lane & 15;                     // anchor column (N)
    const int kb   = (lane >> 4) * 16;              // K sub-range per half-wave
    const int a    = a0 + col;
    const int ac   = a < NA ? a : NA - 1;           // keep EXEC all-ones for WMMA

    v16h ones, bm0, bm1;
#pragma unroll
    for (int j = 0; j < 16; ++j) {
        ones[j] = (_Float16)1.0f;
        bm0[j] = (_Float16)mpos_ws[((size_t)b * NB + (kb + j)) * NA + ac];
        bm1[j] = (_Float16)mpos_ws[((size_t)b * NB + (32 + kb + j)) * NA + ac];
    }
    v8f c = {};
    c = __builtin_amdgcn_wmma_f32_16x16x32_f16(false, ones, false, bm0,
                                               (short)0, c, false, false);
    c = __builtin_amdgcn_wmma_f32_16x16x32_f16(false, ones, false, bm1,
                                               (short)0, c, false, false);

    if (lane < 16 && a < NA) {
        float fg = c[0];  // D row 0: column sum for anchor a
        int tg = 0; float fgout = 0.f;
        if (fg > 1.5f) {  // multi-assigned: keep GT with max overlap
            float bv = -1e30f; int bi = 0;
            for (int n = 0; n < NB; ++n) {
                float ov = ovl_ws[((size_t)b * NB + n) * NA + a];
                if (ov > bv) { bv = ov; bi = n; }
            }
            for (int n = 0; n < NB; ++n)
                mpos_ws[((size_t)b * NB + n) * NA + a] = (n == bi) ? 1.f : 0.f;
            tg = bi; fgout = 1.f;
        } else {
            int first = -1;
            for (int n = 0; n < NB; ++n) {
                if (first < 0 && mpos_ws[((size_t)b * NB + n) * NA + a] > 0.5f)
                    first = n;
            }
            tg = (first < 0) ? 0 : first;
            fgout = (fg > 0.5f) ? 1.f : 0.f;
        }
        const size_t o = (size_t)b * NA + a;
        fg_ws[o]  = fgout;
        tgi_ws[o] = tg;
        out_fg[o]  = fgout;
        out_tgi[o] = (float)tg;
        const float* gb = gt_bboxes + ((size_t)b * NB + tg) * 4;
        out_tb[o * 4 + 0] = gb[0];
        out_tb[o * 4 + 1] = gb[1];
        out_tb[o * 4 + 2] = gb[2];
        out_tb[o * 4 + 3] = gb[3];
    }
}

// ---------------------------------------------------------------------------
// K4: per (b, gt): pos_align = max_a align*mask, pos_overlaps = max_a ovl*mask
// ---------------------------------------------------------------------------
__global__ void k4_pos(const float* __restrict__ align_ws,
                       const float* __restrict__ ovl_ws,
                       const float* __restrict__ mpos_ws,
                       float* __restrict__ pos_a,
                       float* __restrict__ pos_o) {
    __shared__ float sa[256], so[256];
    const int b = blockIdx.x, n = blockIdx.y, tid = threadIdx.x;
    const size_t base = ((size_t)b * NB + n) * NA;
    float ma = 0.f, mo = 0.f;
    for (int a = tid; a < NA; a += 256) {
        float mp = mpos_ws[base + a];
        if (mp > 0.f) {
            ma = fmaxf(ma, align_ws[base + a] * mp);
            mo = fmaxf(mo, ovl_ws[base + a] * mp);
        }
    }
    sa[tid] = ma; so[tid] = mo;
    __syncthreads();
    for (int s = 128; s > 0; s >>= 1) {
        if (tid < s) {
            sa[tid] = fmaxf(sa[tid], sa[tid + s]);
            so[tid] = fmaxf(so[tid], so[tid + s]);
        }
        __syncthreads();
    }
    if (tid == 0) { pos_a[b * NB + n] = sa[0]; pos_o[b * NB + n] = so[0]; }
}

// ---------------------------------------------------------------------------
// K5: per anchor: norm = max_n align*mask * pos_o[n]/(pos_a[n]+eps);
// write target_scores (coalesced 80-wide one-hot writes).
// ---------------------------------------------------------------------------
__global__ void k5_scores(const float* __restrict__ pos_a,
                          const float* __restrict__ pos_o,
                          const float* __restrict__ align_ws,
                          const float* __restrict__ mpos_ws,
                          const int*   __restrict__ tgi_ws,
                          const float* __restrict__ fg_ws,
                          const int*   __restrict__ gt_labels,
                          float* __restrict__ out_ts) {
    __shared__ float spa[NB], spo[NB];
    __shared__ float snorm[256];
    __shared__ int   slab[256];
    __shared__ float sfg[256];
    const int b = blockIdx.x, tid = threadIdx.x;
    const int a0 = blockIdx.y * 256;
    if (tid < NB) { spa[tid] = pos_a[b * NB + tid]; spo[tid] = pos_o[b * NB + tid]; }
    __syncthreads();
    const int a = a0 + tid;
    float norm = 0.f, fg = 0.f; int lab = 0;
    if (a < NA) {
        for (int n = 0; n < NB; ++n) {
            const size_t idx = ((size_t)b * NB + n) * NA + a;
            float mp = mpos_ws[idx];
            if (mp > 0.f) {
                float cand = align_ws[idx] * mp * spo[n] / (spa[n] + 1e-9f);
                norm = fmaxf(norm, cand);
            }
        }
        int tg = tgi_ws[(size_t)b * NA + a];
        lab = gt_labels[b * NB + tg];
        fg  = fg_ws[(size_t)b * NA + a];
    }
    snorm[tid] = norm; slab[tid] = lab; sfg[tid] = fg;
    __syncthreads();
    for (int i = tid; i < 256 * NC; i += 256) {
        const int al = i / NC, cc = i % NC;
        const int aa = a0 + al;
        if (aa < NA)
            out_ts[((size_t)b * NA + aa) * NC + cc] =
                (sfg[al] > 0.f && cc == slab[al]) ? snorm[al] : 0.f;
    }
}

// ---------------------------------------------------------------------------
extern "C" void kernel_launch(void* const* d_in, const int* in_sizes, int n_in,
                              void* d_out, int out_size, void* d_ws, size_t ws_size,
                              hipStream_t stream) {
    const float* pd_scores  = (const float*)d_in[0];
    const float* pd_bboxes  = (const float*)d_in[1];
    const float* anc_points = (const float*)d_in[2];
    const int*   gt_labels  = (const int*)  d_in[3];
    const float* gt_bboxes  = (const float*)d_in[4];
    const float* mask_gt    = (const float*)d_in[5];

    const size_t MAT = (size_t)BS * NB * NA;   // 8,601,600 floats per matrix
    float* ws       = (float*)d_ws;
    float* align_ws = ws;
    float* ovl_ws   = ws + MAT;
    float* mpos_ws  = ws + 2 * MAT;
    float* pos_a    = ws + 3 * MAT;
    float* pos_o    = pos_a + (size_t)BS * NB;
    int*   tgi_ws   = (int*)(pos_o + (size_t)BS * NB);
    float* fg_ws    = (float*)(tgi_ws + (size_t)BS * NA);

    float* out     = (float*)d_out;
    float* out_tb  = out;                                 // (bs, na, 4)
    float* out_ts  = out + (size_t)BS * NA * 4;           // (bs, na, nc)
    float* out_fg  = out_ts + (size_t)BS * NA * NC;       // (bs, na)
    float* out_tgi = out_fg + (size_t)BS * NA;            // (bs, na)

    dim3 blk(256);
    k1_metrics<<<dim3(BS, (NA + 255) / 256), blk, 0, stream>>>(
        pd_scores, pd_bboxes, anc_points, gt_labels, gt_bboxes, mask_gt,
        align_ws, ovl_ws);
    k2_topk<<<dim3(BS, NB), blk, 0, stream>>>(
        anc_points, gt_bboxes, mask_gt, align_ws, mpos_ws);
    k3_resolve<<<dim3(BS, (NA + 127) / 128), blk, 0, stream>>>(
        ovl_ws, mpos_ws, gt_bboxes, fg_ws, tgi_ws, out_tb, out_fg, out_tgi);
    k4_pos<<<dim3(BS, NB), blk, 0, stream>>>(
        align_ws, ovl_ws, mpos_ws, pos_a, pos_o);
    k5_scores<<<dim3(BS, (NA + 255) / 256), blk, 0, stream>>>(
        pos_a, pos_o, align_ws, mpos_ws, tgi_ws, fg_ws, gt_labels, out_ts);
}
